// GraphRec_61203283968781
// MI455X (gfx1250) — matrix operand order repaired and checked by reading, not compiled, not run on hardware
//
#include <hip/hip_runtime.h>
#include <hip/hip_bf16.h>

// ---------------------------------------------------------------------------
// GraphRec forward for MI455X (gfx1250, wave32, WMMA).
//
// Roofline: ~560MB of embedding gathers (tables are L2-resident: 25.6MB each
// vs 192MB L2) dominate; ~36 GFLOP of GEMM rides on top via
// v_wmma_f32_16x16x32_bf16. Attention is fully on-chip: one workgroup per
// batch element keeps gathered item rows in LDS (bf16) and reuses them for
// the projection-GEMM A operand and the softmax-weighted sum.
//
// Round-2 changes (driven by the instruction histogram):
//  * weights stored TRANSPOSED in LDS -> B fragments are two 16B
//    spans (2x uint4 loads) instead of 16 strided ds_load_u16.
//  * LDS rows padded (+8 ushorts) to rotate banks across fragment lanes.
//  * all LDS staging is uint4 (b128) packed-bf16; weighted sum reads packed
//    u32 pairs.
// ---------------------------------------------------------------------------

typedef __attribute__((ext_vector_type(16))) __bf16 v16bf;
typedef __attribute__((ext_vector_type(8)))  unsigned int v8u;
typedef __attribute__((ext_vector_type(8)))  float v8f;

#define EDIM     64
#define BATCH    8192
#define MASK_VAL (-100000000.0f)

#define IT_PITCH 72     // sItems row pitch (ushorts): 144B, 16B-aligned, bank-rotating
#define W_PITCH  136    // transposed-weight row pitch (ushorts): 272B

__device__ __forceinline__ unsigned short f2bf(float f) {
    unsigned u = __builtin_bit_cast(unsigned, f);
    unsigned r = u + 0x7FFFu + ((u >> 16) & 1u);   // round-to-nearest-even
    return (unsigned short)(r >> 16);
}
__device__ __forceinline__ float bf2f(unsigned short h) {
    unsigned u = ((unsigned)h) << 16;
    return __builtin_bit_cast(float, u);
}
__device__ __forceinline__ unsigned pk2(float a, float b) {
    return (unsigned)f2bf(a) | ((unsigned)f2bf(b) << 16);
}
// Load a 16x32-slice fragment: 8 bf16 at p, 8 bf16 at p+gap (gap in ushorts).
// A operand: gap=16 (K pairs split 0..7 / 16..23); B operand: gap=16 (same split).
__device__ __forceinline__ v16bf load_frag(const unsigned short* p, int gap) {
    const uint4 lo = *(const uint4*)p;
    const uint4 hi = *(const uint4*)(p + gap);
    v8u t;
    t[0] = lo.x; t[1] = lo.y; t[2] = lo.z; t[3] = lo.w;
    t[4] = hi.x; t[5] = hi.y; t[6] = hi.z; t[7] = hi.w;
    return __builtin_bit_cast(v16bf, t);
}

// ---------------------------------------------------------------------------
// Attention: out[b,:] = sum_l softmax_l(w2.relu(W1.[item_l;user]) + b2 + mask)
//                       * item_l
// One workgroup (256 thr = 8 waves) per batch element.
// ---------------------------------------------------------------------------
__global__ __launch_bounds__(256) void attn_kernel(
    const int*   __restrict__ ids,     // [B, L]
    const float* __restrict__ table,   // item/user embedding table [*,64]
    const int*   __restrict__ user,    // [B]
    const float* __restrict__ utable,  // user embedding table      [*,64]
    const float* __restrict__ w1,      // [128,64]
    const float* __restrict__ b1,      // [64]
    const float* __restrict__ w2,      // [64]
    const float* __restrict__ b2,      // [1]
    float*       __restrict__ outv,    // [B,64]
    int L)
{
    __shared__ unsigned short sItems[208][IT_PITCH];      // gathered rows, bf16
    __shared__ unsigned short sW1t[64][W_PITCH];          // W1 transposed [n][k]
    __shared__ __align__(16) unsigned short sUser[64];    // query user emb
    __shared__ int   sId[208];
    __shared__ float sLogit[208];
    __shared__ float sMask[208];
    __shared__ float sRed[256];
    __shared__ float sPart[8][64];

    const int b    = blockIdx.x;
    const int tid  = threadIdx.x;
    const int lane = tid & 31;
    const int wave = tid >> 5;
    const int Lp   = (L + 15) & ~15;
    const int nMT  = Lp >> 4;

    // --- stage ids + query user embedding ---
    for (int r = tid; r < Lp; r += 256) sId[r] = (r < L) ? ids[b * L + r] : 0;
    if (tid < 64) {
        const int uid = user[b];
        sUser[tid] = f2bf(utable[(size_t)uid * EDIM + tid]);
    }
    // --- stage W1 transposed: sW1t[n][k] = w1[k*64+n], 8 k per thread ---
    for (int i = tid; i < 64 * 16; i += 256) {
        int n = i >> 4, kc = (i & 15) * 8;
        uint4 q;
        q.x = pk2(w1[(kc + 0) * 64 + n], w1[(kc + 1) * 64 + n]);
        q.y = pk2(w1[(kc + 2) * 64 + n], w1[(kc + 3) * 64 + n]);
        q.z = pk2(w1[(kc + 4) * 64 + n], w1[(kc + 5) * 64 + n]);
        q.w = pk2(w1[(kc + 6) * 64 + n], w1[(kc + 7) * 64 + n]);
        *(uint4*)&sW1t[n][kc] = q;
    }
    __syncthreads();   // sId ready

    // --- gather item rows (8 floats per thread, float4 loads, b128 stores) ---
    for (int i = tid; i < Lp * 8; i += 256) {
        int r = i >> 3, cc = (i & 7) * 8;
        uint4 q = {0u, 0u, 0u, 0u};
        if (r < L) {
            const float4* t4 = (const float4*)(table + (size_t)sId[r] * EDIM + cc);
            float4 f0 = t4[0], f1 = t4[1];
            q.x = pk2(f0.x, f0.y); q.y = pk2(f0.z, f0.w);
            q.z = pk2(f1.x, f1.y); q.w = pk2(f1.z, f1.w);
        }
        *(uint4*)&sItems[r][cc] = q;
    }
    for (int r = tid; r < Lp; r += 256)
        sMask[r] = (r < L) ? ((sId[r] == 0) ? MASK_VAL : 0.0f) : -3.0e38f;
    __syncthreads();

    // --- projection GEMM + fused logit dot, per wave ---
    const int mloc  = lane & 15;
    const int khalf = (lane & 16) ? 8 : 0;    // A K sub-offset, lanes 16-31
    const int kb2   = (lane & 16) ? 8 : 0;    // B K sub-offset, lanes 16-31
    const int ncol  = lane & 15;

    for (int mt = wave; mt < nMT; mt += 8) {
        const int m = mt * 16 + mloc;
        float lsum[8];
        #pragma unroll
        for (int v = 0; v < 8; ++v) lsum[v] = 0.0f;

        // A fragments (4 K-slices): slices 0,1 from items; 2,3 from user emb
        v16bf afr[4];
        #pragma unroll
        for (int kt = 0; kt < 4; ++kt) {
            const int kbase = kt * 32;
            const unsigned short* ap = (kbase < 64)
                ? &sItems[m][kbase + khalf]
                : &sUser[kbase - 64 + khalf];
            afr[kt] = load_frag(ap, 16);
        }

        #pragma unroll
        for (int nt = 0; nt < 4; ++nt) {
            v8f acc = {};
            const unsigned short* brow = &sW1t[nt * 16 + ncol][kb2];
            #pragma unroll
            for (int kt = 0; kt < 4; ++kt) {
                v16bf bfr = load_frag(brow + kt * 32, 16);
                acc = __builtin_amdgcn_wmma_f32_16x16x32_bf16(
                        false, afr[kt], false, bfr, (short)0, acc, false, false);
            }
            const int n    = nt * 16 + ncol;
            const float bn  = b1[n];
            const float w2n = w2[n];
            #pragma unroll
            for (int v = 0; v < 8; ++v) {
                float h = acc[v] + bn;
                h = h > 0.0f ? h : 0.0f;
                lsum[v] += h * w2n;
            }
        }
        #pragma unroll
        for (int off = 1; off < 16; off <<= 1) {
            #pragma unroll
            for (int v = 0; v < 8; ++v)
                lsum[v] += __shfl_xor(lsum[v], off, 32);
        }
        if ((lane & 15) == 0) {
            const int mrow = mt * 16 + ((lane & 16) ? 8 : 0);
            #pragma unroll
            for (int v = 0; v < 8; ++v)
                sLogit[mrow + v] = lsum[v] + b2[0] + sMask[mrow + v];
        }
    }
    __syncthreads();

    // --- softmax (padded rows carry -3e38 -> weight 0) ---
    float lmax = -3.0e38f;
    for (int r = tid; r < Lp; r += 256) lmax = fmaxf(lmax, sLogit[r]);
    sRed[tid] = lmax;
    __syncthreads();
    for (int s = 128; s > 0; s >>= 1) {
        if (tid < s) sRed[tid] = fmaxf(sRed[tid], sRed[tid + s]);
        __syncthreads();
    }
    const float gmax = sRed[0];
    __syncthreads();

    float lacc = 0.0f;
    for (int r = tid; r < Lp; r += 256) {
        float e = __expf(sLogit[r] - gmax);
        sLogit[r] = e;
        lacc += e;
    }
    sRed[tid] = lacc;
    __syncthreads();
    for (int s = 128; s > 0; s >>= 1) {
        if (tid < s) sRed[tid] += sRed[tid + s];
        __syncthreads();
    }
    const float ginv = 1.0f / sRed[0];

    // --- weighted sum: packed u32 reads (2 bf16/load), 8 l-strided groups ---
    {
        const int cp  = tid & 31;       // column pair: cols 2cp, 2cp+1
        const int grp = tid >> 5;       // 8 row groups
        float a0 = 0.0f, a1 = 0.0f;
        for (int r = grp; r < Lp; r += 8) {
            const unsigned u = *(const unsigned*)&sItems[r][2 * cp];
            const float wgt = sLogit[r];
            a0 += wgt * bf2f((unsigned short)(u & 0xFFFFu));
            a1 += wgt * bf2f((unsigned short)(u >> 16));
        }
        sPart[grp][2 * cp]     = a0;
        sPart[grp][2 * cp + 1] = a1;
    }
    __syncthreads();
    if (tid < 64) {
        float s = 0.0f;
        #pragma unroll
        for (int g = 0; g < 8; ++g) s += sPart[g][tid];
        outv[(size_t)b * EDIM + tid] = s * ginv;
    }
}

// ---------------------------------------------------------------------------
// Generic 64-rows-per-block WMMA MLP stage:
//   out[m,:64] = act( concat(gather(a0), gather(a1))[m,:K] @ W[K,64] + bias )
// idx == nullptr -> dense row m; else row idx[m]. a1 used only when K == 128.
// ---------------------------------------------------------------------------
__global__ __launch_bounds__(128) void mlp64_kernel(
    const float* __restrict__ a0, const int* __restrict__ idx0,
    const float* __restrict__ a1, const int* __restrict__ idx1,
    const float* __restrict__ W,  const float* __restrict__ bias,
    float* __restrict__ out, int K, int doRelu)
{
    __shared__ unsigned short sA[64][W_PITCH];    // concat rows, bf16
    __shared__ unsigned short sWt[64][W_PITCH];   // W transposed [n][k]
    __shared__ int sRid0[64];
    __shared__ int sRid1[64];

    const int tid     = threadIdx.x;
    const int lane    = tid & 31;
    const int wave    = tid >> 5;           // 4 waves -> 4 M-tiles of 16 rows
    const int rowBase = blockIdx.x * 64;

    if (tid < 64) {
        const int gr = rowBase + tid;
        sRid0[tid] = idx0 ? idx0[gr] : gr;
        sRid1[tid] = (K == 128) ? (idx1 ? idx1[gr] : gr) : 0;
    }
    // --- stage W transposed: sWt[n][k] = W[k*64+n] ---
    for (int i = tid; i < 64 * (K >> 3); i += 128) {
        int n = i / (K >> 3), kc = (i % (K >> 3)) * 8;
        uint4 q;
        q.x = pk2(W[(kc + 0) * 64 + n], W[(kc + 1) * 64 + n]);
        q.y = pk2(W[(kc + 2) * 64 + n], W[(kc + 3) * 64 + n]);
        q.z = pk2(W[(kc + 4) * 64 + n], W[(kc + 5) * 64 + n]);
        q.w = pk2(W[(kc + 6) * 64 + n], W[(kc + 7) * 64 + n]);
        *(uint4*)&sWt[n][kc] = q;
    }
    __syncthreads();   // sRid ready

    // --- stage A rows: 8 cols per thread, float4 gathers, b128 stores ---
    for (int i = tid; i < 64 * (K >> 3); i += 128) {
        int r = i / (K >> 3), cc = (i % (K >> 3)) * 8;
        const float* src = (cc < 64)
            ? (a0 + (size_t)sRid0[r] * 64 + cc)
            : (a1 + (size_t)sRid1[r] * 64 + (cc - 64));
        const float4* t4 = (const float4*)src;
        float4 f0 = t4[0], f1 = t4[1];
        uint4 q;
        q.x = pk2(f0.x, f0.y); q.y = pk2(f0.z, f0.w);
        q.z = pk2(f1.x, f1.y); q.w = pk2(f1.z, f1.w);
        *(uint4*)&sA[r][cc] = q;
    }
    __syncthreads();

    const int m      = wave * 16 + (lane & 15);
    const int khalf  = (lane & 16) ? 8 : 0;
    const int kb2    = (lane & 16) ? 8 : 0;
    const int ncol   = lane & 15;
    const int kTiles = K >> 5;

    // A fragments once per M-tile, reused over all 4 N-tiles
    v16bf afr[4];
    for (int kt = 0; kt < kTiles; ++kt)
        afr[kt] = load_frag(&sA[m][kt * 32 + khalf], 16);

    #pragma unroll
    for (int nt = 0; nt < 4; ++nt) {
        v8f acc = {};
        const unsigned short* brow = &sWt[nt * 16 + ncol][kb2];
        for (int kt = 0; kt < kTiles; ++kt) {
            v16bf bfr = load_frag(brow + kt * 32, 16);
            acc = __builtin_amdgcn_wmma_f32_16x16x32_bf16(
                    false, afr[kt], false, bfr, (short)0, acc, false, false);
        }
        const int n  = nt * 16 + ncol;
        const float bn = bias[n];
        #pragma unroll
        for (int v = 0; v < 8; ++v) {
            float x = acc[v] + bn;
            if (doRelu) x = x > 0.0f ? x : 0.0f;
            int mrow = rowBase + wave * 16 + v + ((lane & 16) ? 8 : 0);
            out[(size_t)mrow * 64 + n] = x;
        }
    }
}

// Final [64] -> 1 projection (rp3): bandwidth-trivial scalar dot, float4 loads.
__global__ __launch_bounds__(256) void dot_kernel(
    const float* __restrict__ x, const float* __restrict__ w,
    const float* __restrict__ b, float* __restrict__ out, int n)
{
    int i = blockIdx.x * blockDim.x + threadIdx.x;
    if (i >= n) return;
    const float4* xv = (const float4*)(x + (size_t)i * 64);
    const float4* wv = (const float4*)w;
    float acc = b[0];
    #pragma unroll
    for (int k = 0; k < 16; ++k) {
        float4 xf = xv[k], wf = wv[k];
        acc += xf.x * wf.x + xf.y * wf.y + xf.z * wf.z + xf.w * wf.w;
    }
    out[i] = acc;
}

extern "C" void kernel_launch(void* const* d_in, const int* in_sizes, int n_in,
                              void* d_out, int out_size, void* d_ws, size_t ws_size,
                              hipStream_t stream) {
    (void)in_sizes; (void)n_in; (void)out_size; (void)ws_size;
    const int*   user      = (const int*)d_in[0];
    const int*   user_hist = (const int*)d_in[1];
    const int*   user_nbrs = (const int*)d_in[2];
    const int*   pos_item  = (const int*)d_in[3];
    const int*   neg_item  = (const int*)d_in[4];
    const float* ue        = (const float*)d_in[5];
    const float* ie        = (const float*)d_in[6];
    const float* ia_w1 = (const float*)d_in[7],  *ia_b1 = (const float*)d_in[8];
    const float* ia_w2 = (const float*)d_in[9],  *ia_b2 = (const float*)d_in[10];
    const float* ua_w1 = (const float*)d_in[11], *ua_b1 = (const float*)d_in[12];
    const float* ua_w2 = (const float*)d_in[13], *ua_b2 = (const float*)d_in[14];
    const float* fuse_w = (const float*)d_in[15], *fuse_b = (const float*)d_in[16];
    const float* self_w = (const float*)d_in[17], *self_b = (const float*)d_in[18];
    const float* ul1_w = (const float*)d_in[19], *ul1_b = (const float*)d_in[20];
    const float* ul2_w = (const float*)d_in[21], *ul2_b = (const float*)d_in[22];
    const float* il1_w = (const float*)d_in[23], *il1_b = (const float*)d_in[24];
    const float* il2_w = (const float*)d_in[25], *il2_b = (const float*)d_in[26];
    const float* rp1_w = (const float*)d_in[27], *rp1_b = (const float*)d_in[28];
    const float* rp2_w = (const float*)d_in[29], *rp2_b = (const float*)d_in[30];
    const float* rp3_w = (const float*)d_in[31], *rp3_b = (const float*)d_in[32];
    float* out = (float*)d_out;

    const int Bn = BATCH;
    float* ws       = (float*)d_ws;
    float* h_item   = ws + (size_t)0 * Bn * 64;
    float* h_social = ws + (size_t)1 * Bn * 64;
    float* hbuf     = ws + (size_t)2 * Bn * 64;
    float* hu0      = ws + (size_t)3 * Bn * 64;
    float* tbuf     = ws + (size_t)4 * Bn * 64;
    float* hu       = ws + (size_t)5 * Bn * 64;
    float* pos_hi   = ws + (size_t)6 * Bn * 64;
    float* neg_hi   = ws + (size_t)7 * Bn * 64;
    float* r1       = ws + (size_t)8 * Bn * 64;
    float* r2       = ws + (size_t)9 * Bn * 64;

    // user-encoder attention blocks (the gather/GEMM heavy part)
    attn_kernel<<<Bn, 256, 0, stream>>>(user_hist, ie, user, ue,
                                        ia_w1, ia_b1, ia_w2, ia_b2, h_item, 200);
    attn_kernel<<<Bn, 256, 0, stream>>>(user_nbrs, ue, user, ue,
                                        ua_w1, ua_b1, ua_w2, ua_b2, h_social, 64);

    // h = relu([h_item; h_social] @ fuse_w + fuse_b)
    mlp64_kernel<<<Bn / 64, 128, 0, stream>>>(h_item, nullptr, h_social, nullptr,
                                              fuse_w, fuse_b, hbuf, 128, 1);
    // hu0 = [h; user_emb] @ self_w + self_b
    mlp64_kernel<<<Bn / 64, 128, 0, stream>>>(hbuf, nullptr, ue, user,
                                              self_w, self_b, hu0, 128, 0);
    // hu = relu(hu0 @ ul1) @ ul2
    mlp64_kernel<<<Bn / 64, 128, 0, stream>>>(hu0, nullptr, nullptr, nullptr,
                                              ul1_w, ul1_b, tbuf, 64, 1);
    mlp64_kernel<<<Bn / 64, 128, 0, stream>>>(tbuf, nullptr, nullptr, nullptr,
                                              ul2_w, ul2_b, hu, 64, 0);
    // pos/neg item towers (gathered straight from the item table)
    mlp64_kernel<<<Bn / 64, 128, 0, stream>>>(ie, pos_item, nullptr, nullptr,
                                              il1_w, il1_b, tbuf, 64, 1);
    mlp64_kernel<<<Bn / 64, 128, 0, stream>>>(tbuf, nullptr, nullptr, nullptr,
                                              il2_w, il2_b, pos_hi, 64, 0);
    mlp64_kernel<<<Bn / 64, 128, 0, stream>>>(ie, neg_item, nullptr, nullptr,
                                              il1_w, il1_b, tbuf, 64, 1);
    mlp64_kernel<<<Bn / 64, 128, 0, stream>>>(tbuf, nullptr, nullptr, nullptr,
                                              il2_w, il2_b, neg_hi, 64, 0);

    // rate predictor, positive
    mlp64_kernel<<<Bn / 64, 128, 0, stream>>>(hu, nullptr, pos_hi, nullptr,
                                              rp1_w, rp1_b, r1, 128, 1);
    mlp64_kernel<<<Bn / 64, 128, 0, stream>>>(r1, nullptr, nullptr, nullptr,
                                              rp2_w, rp2_b, r2, 64, 1);
    dot_kernel<<<(Bn + 255) / 256, 256, 0, stream>>>(r2, rp3_w, rp3_b, out, Bn);

    // rate predictor, negative
    mlp64_kernel<<<Bn / 64, 128, 0, stream>>>(hu, nullptr, neg_hi, nullptr,
                                              rp1_w, rp1_b, r1, 128, 1);
    mlp64_kernel<<<Bn / 64, 128, 0, stream>>>(r1, nullptr, nullptr, nullptr,
                                              rp2_w, rp2_b, r2, 64, 1);
    dot_kernel<<<(Bn + 255) / 256, 256, 0, stream>>>(r2, rp3_w, rp3_b, out + Bn, Bn);
}
